// SATLayer_22763326669382
// MI455X (gfx1250) — matrix operand order repaired and checked by reading, not compiled, not run on hardware
//
#include <hip/hip_runtime.h>
#include <hip/hip_bf16.h>

// ---------------------------------------------------------------------------
// CDNA5 (gfx1250) SAT layer: TransformerConv + subgraph MLP, bf16 WMMA GEMMs.
// Round 2: wave-uniform fast-path epilogues (no per-element saveexec chains).
// ---------------------------------------------------------------------------

typedef __attribute__((ext_vector_type(16))) __bf16 v16bf;
typedef __attribute__((ext_vector_type(8)))  float  v8f;

union Frag16 { v16bf v; uint4 q[2]; };

__device__ __forceinline__ unsigned short f2bf(float f) {
  unsigned u = __float_as_uint(f);
  u += 0x7fffu + ((u >> 16) & 1u);           // round-to-nearest-even
  return (unsigned short)(u >> 16);
}

// monotonic uint key for float atomicMax (segment-max over edges)
__device__ __forceinline__ unsigned fkey(float f) {
  unsigned u = __float_as_uint(f);
  return (u & 0x80000000u) ? ~u : (u | 0x80000000u);
}
__device__ __forceinline__ float funkey(unsigned k) {
  return (k & 0x80000000u) ? __uint_as_float(k & 0x7fffffffu)
                           : __uint_as_float(~k);
}

// A-fragment (16x32 bf16) from an LDS tile [16][rowStride] (row-major, bf16).
// ISA 7.12.2: lane m<16 -> M=m, K pairs {hi*8..hi*8+7} and {16+hi*8..+7}.
__device__ __forceinline__ v16bf load_a_lds(const unsigned short* tile,
                                            int rowStride, int lane, int ks) {
  int m = lane & 15, hi = lane >> 4;
  const unsigned short* base = tile + m * rowStride + ks * 32 + hi * 8;
  Frag16 f;
  f.q[0] = *reinterpret_cast<const uint4*>(base);
  f.q[1] = *reinterpret_cast<const uint4*>(base + 16);
  return f.v;
}

// B-fragment (32x16 bf16) from transposed bf16 weights WT[n][k] (k contiguous).
// lane n<16 -> N=n holds K {hi*8..hi*8+7} and {16+hi*8..+7} (matches A).
__device__ __forceinline__ v16bf load_b_glb(const unsigned short* wt, int Kdim,
                                            int lane, int nt, int ks) {
  int n = lane & 15, hi = lane >> 4;
  const unsigned short* base = wt + (nt * 16 + n) * Kdim + ks * 32 + hi * 8;
  Frag16 f;
  f.q[0] = *reinterpret_cast<const uint4*>(base);
  f.q[1] = *reinterpret_cast<const uint4*>(base + 16);
  return f.v;
}

#define WMMA_BF16(A, B, C) \
  __builtin_amdgcn_wmma_f32_16x16x32_bf16(false, (A), false, (B), (short)0, (C), false, false)

// ---------------------------------------------------------------------------
__global__ void zero_kernel(float* __restrict__ p, long n) {
  long i = (long)blockIdx.x * blockDim.x + threadIdx.x;
  long st = (long)gridDim.x * blockDim.x;
  for (; i < n; i += st) p[i] = 0.0f;
}

// Transpose + convert all weight matrices to bf16 (WT[n][k] layout).
__global__ void prep_weights_kernel(const float* __restrict__ Wq, const float* __restrict__ Wk,
                                    const float* __restrict__ Wv, const float* __restrict__ Ws,
                                    const float* __restrict__ We, const float* __restrict__ P1,
                                    const float* __restrict__ P2, unsigned short* __restrict__ w) {
  int i = blockIdx.x * blockDim.x + threadIdx.x;
  int st = gridDim.x * blockDim.x;
  for (int t = i; t < 16384; t += st) {          // 128x128 transposes
    int nn = t >> 7, kk = t & 127;
    float a = Wq[kk * 128 + nn], b = Wk[kk * 128 + nn];
    float c = Wv[kk * 128 + nn], d = Ws[kk * 128 + nn];
    float e = P1[kk * 128 + nn], f = P2[kk * 128 + nn];
    w[t]              = f2bf(a);
    w[16384 + t]      = f2bf(b);
    w[32768 + t]      = f2bf(c);
    w[49152 + t]      = f2bf(d);
    w[73728 + t]      = f2bf(e);
    w[90112 + t]      = f2bf(f);
  }
  for (int t = i; t < 8192; t += st) {           // We: [64 in][128 out] -> WT[128][64]
    int nn = t >> 6, kk = t & 63;
    w[65536 + t] = f2bf(We[kk * 128 + nn]);
  }
}

// ---------------------------------------------------------------------------
// q/k/v/skip projections: block = 4 waves = 64 rows; each wave does 16 rows.
__global__ __launch_bounds__(128) void node_proj_kernel(
    const float* __restrict__ x, const unsigned short* __restrict__ wts,
    const float* __restrict__ bq, const float* __restrict__ bk,
    const float* __restrict__ bv, const float* __restrict__ bs,
    float* __restrict__ qo, float* __restrict__ ko,
    float* __restrict__ vo, float* __restrict__ so, int n) {
  __shared__ __align__(16) unsigned short xt[64 * 128];
  int tid = threadIdx.x, wave = tid >> 5, lane = tid & 31;
  int row0 = blockIdx.x * 64;
  for (int t = tid; t < 64 * 128; t += 128) {
    int r = t >> 7, c = t & 127;
    int gr = row0 + r;
    float f = (gr < n) ? x[gr * 128 + c] : 0.0f;
    xt[t] = f2bf(f);
  }
  if ((row0 + 64) * 128 < n * 128)
    __builtin_prefetch(&x[(row0 + 64) * 128 + tid], 0, 1);
  __syncthreads();

  const unsigned short* mytile = xt + wave * 16 * 128;
  v16bf a[4];
#pragma unroll
  for (int ks = 0; ks < 4; ++ks) a[ks] = load_a_lds(mytile, 128, lane, ks);

  int m = lane & 15, hi = lane >> 4;
  bool fullw = (row0 + wave * 16 + 16) <= n;     // wave-uniform tile-in-bounds
  const float* biases[4] = {bq, bk, bv, bs};
  float* outs[4] = {qo, ko, vo, so};
#pragma unroll
  for (int wmat = 0; wmat < 4; ++wmat) {
    const unsigned short* wt = wts + wmat * 16384;
    const float* bias = biases[wmat];
    float* out = outs[wmat];
#pragma unroll
    for (int nt = 0; nt < 8; ++nt) {
      v8f c = {};
#pragma unroll
      for (int ks = 0; ks < 4; ++ks) {
        v16bf b = load_b_glb(wt, 128, lane, nt, ks);
        c = WMMA_BF16(a[ks], b, c);
      }
      int col = nt * 16 + m;
      float bb = bias[col];
      int base = row0 + wave * 16 + 8 * hi;
      if (fullw) {
#pragma unroll
        for (int r = 0; r < 8; ++r) out[(base + r) * 128 + col] = c[r] + bb;
      } else {
#pragma unroll
        for (int r = 0; r < 8; ++r) {
          int grow = base + r;
          if (grow < n) out[grow * 128 + col] = c[r] + bb;
        }
      }
    }
  }
}

// ---------------------------------------------------------------------------
// Edge pass A: e = edge_attr@We+be (WMMA), alpha = q[dst].(k[src]+e)/4,
// segment-max via atomicMax on ordered keys.
__global__ __launch_bounds__(128) void edge_attn_a_kernel(
    const float* __restrict__ ea, const int* __restrict__ src, const int* __restrict__ dst,
    const unsigned short* __restrict__ weT, const float* __restrict__ be,
    const float* __restrict__ qb, const float* __restrict__ kb,
    float* __restrict__ alpha, unsigned* __restrict__ mkey, int E) {
  __shared__ __align__(16) unsigned short et[64 * 64];
  int tid = threadIdx.x, wave = tid >> 5, lane = tid & 31;
  int e0 = blockIdx.x * 64;
  for (int t = tid; t < 64 * 64; t += 128) {
    int r = t >> 6, c = t & 63;
    float f = (e0 + r < E) ? ea[(e0 + r) * 64 + c] : 0.0f;
    et[t] = f2bf(f);
  }
  __builtin_prefetch(&ea[(long)(e0 + 64) * 64 + tid], 0, 1);
  __syncthreads();

  const unsigned short* mytile = et + wave * 16 * 64;
  v16bf a[2];
#pragma unroll
  for (int ks = 0; ks < 2; ++ks) a[ks] = load_a_lds(mytile, 64, lane, ks);
  int m = lane & 15, hi = lane >> 4;
  int we0 = e0 + wave * 16;
  bool fullw = (we0 + 16) <= E;                  // wave-uniform
#pragma unroll
  for (int h = 0; h < 8; ++h) {
    v8f c = {};
#pragma unroll
    for (int ks = 0; ks < 2; ++ks) {
      v16bf b = load_b_glb(weT, 64, lane, h, ks);
      c = WMMA_BF16(a[ks], b, c);
    }
    float bb = be[h * 16 + m];
    if (fullw) {
#pragma unroll
      for (int r = 0; r < 8; ++r) {
        int eg = we0 + r + 8 * hi;
        int s = src[eg], d = dst[eg];
        float ev = c[r] + bb;
        float kv = kb[s * 128 + h * 16 + m];
        float qv = qb[d * 128 + h * 16 + m];
        float p = qv * (kv + ev);
        p += __shfl_xor(p, 1, 16);
        p += __shfl_xor(p, 2, 16);
        p += __shfl_xor(p, 4, 16);
        p += __shfl_xor(p, 8, 16);
        p *= 0.25f;                              // 1/sqrt(C), C=16
        if (m == 0) {
          alpha[eg * 8 + h] = p;
          atomicMax(&mkey[d * 8 + h], fkey(p));
        }
      }
    } else {
#pragma unroll
      for (int r = 0; r < 8; ++r) {
        int eg = we0 + r + 8 * hi;
        if (eg < E) {
          int s = src[eg], d = dst[eg];
          float ev = c[r] + bb;
          float kv = kb[s * 128 + h * 16 + m];
          float qv = qb[d * 128 + h * 16 + m];
          float p = qv * (kv + ev);
          p += __shfl_xor(p, 1, 16);
          p += __shfl_xor(p, 2, 16);
          p += __shfl_xor(p, 4, 16);
          p += __shfl_xor(p, 8, 16);
          p *= 0.25f;
          if (m == 0) {
            alpha[eg * 8 + h] = p;
            atomicMax(&mkey[d * 8 + h], fkey(p));
          }
        }
      }
    }
  }
}

// Edge pass B: recompute e, ea=exp(alpha-m), scatter num/denom atomics.
__global__ __launch_bounds__(128) void edge_attn_b_kernel(
    const float* __restrict__ ea, const int* __restrict__ src, const int* __restrict__ dst,
    const unsigned short* __restrict__ weT, const float* __restrict__ be,
    const float* __restrict__ vb, const float* __restrict__ alpha,
    const unsigned* __restrict__ mkey, float* __restrict__ numb,
    float* __restrict__ denb, int E) {
  __shared__ __align__(16) unsigned short et[64 * 64];
  int tid = threadIdx.x, wave = tid >> 5, lane = tid & 31;
  int e0 = blockIdx.x * 64;
  for (int t = tid; t < 64 * 64; t += 128) {
    int r = t >> 6, c = t & 63;
    float f = (e0 + r < E) ? ea[(e0 + r) * 64 + c] : 0.0f;
    et[t] = f2bf(f);
  }
  __syncthreads();

  const unsigned short* mytile = et + wave * 16 * 64;
  v16bf a[2];
#pragma unroll
  for (int ks = 0; ks < 2; ++ks) a[ks] = load_a_lds(mytile, 64, lane, ks);
  int m = lane & 15, hi = lane >> 4;
  int we0 = e0 + wave * 16;
  bool fullw = (we0 + 16) <= E;                  // wave-uniform
#pragma unroll
  for (int h = 0; h < 8; ++h) {
    v8f c = {};
#pragma unroll
    for (int ks = 0; ks < 2; ++ks) {
      v16bf b = load_b_glb(weT, 64, lane, h, ks);
      c = WMMA_BF16(a[ks], b, c);
    }
    float bb = be[h * 16 + m];
    if (fullw) {
#pragma unroll
      for (int r = 0; r < 8; ++r) {
        int eg = we0 + r + 8 * hi;
        int s = src[eg], d = dst[eg];
        float ev = c[r] + bb;
        float al = alpha[eg * 8 + h];
        float mm = funkey(mkey[d * 8 + h]);
        float ex = __expf(al - mm);
        float vv = vb[s * 128 + h * 16 + m];
        atomicAdd(&numb[d * 128 + h * 16 + m], ex * (vv + ev));
        if (m == 0) atomicAdd(&denb[d * 8 + h], ex);
      }
    } else {
#pragma unroll
      for (int r = 0; r < 8; ++r) {
        int eg = we0 + r + 8 * hi;
        if (eg < E) {
          int s = src[eg], d = dst[eg];
          float ev = c[r] + bb;
          float al = alpha[eg * 8 + h];
          float mm = funkey(mkey[d * 8 + h]);
          float ex = __expf(al - mm);
          float vv = vb[s * 128 + h * 16 + m];
          atomicAdd(&numb[d * 128 + h * 16 + m], ex * (vv + ev));
          if (m == 0) atomicAdd(&denb[d * 8 + h], ex);
        }
      }
    }
  }
}

// ---------------------------------------------------------------------------
// Subgraph branch: gather -> LN -> GEMM1 -> SiLU -> GEMM2 -> scatter-add.
__global__ __launch_bounds__(128) void subgraph_kernel(
    const float* __restrict__ x, const int* __restrict__ snodes,
    const int* __restrict__ sind, const float* __restrict__ sng,
    const float* __restrict__ snb, const unsigned short* __restrict__ p1T,
    const float* __restrict__ b1, const unsigned short* __restrict__ p2T,
    const float* __restrict__ b2, float* __restrict__ subacc,
    float* __restrict__ cnt, int M) {
  __shared__ __align__(16) float          xf[4][16 * 128];
  __shared__ __align__(16) unsigned short xb[4][16 * 128];
  __shared__ __align__(16) unsigned short hb[4][16 * 128];
  int tid = threadIdx.x, wave = tid >> 5, lane = tid & 31;
  int m0 = blockIdx.x * 64 + wave * 16;
  bool fullw = (m0 + 16) <= M;                   // wave-uniform

  for (int t = lane; t < 16 * 128; t += 32) {    // gather 16 rows
    int r = t >> 7, c = t & 127;
    int ri = m0 + r;
    int node = (fullw || ri < M) ? snodes[ri < M ? ri : 0] : 0;
    xf[wave][t] = x[node * 128 + c];
  }
  __syncthreads();

  if (lane < 16) {                                // LayerNorm, one lane per row
    const float* rp = &xf[wave][lane * 128];
    unsigned short* wp = &xb[wave][lane * 128];
    float s = 0.0f;
    for (int c = 0; c < 128; ++c) s += rp[c];
    float mu = s * (1.0f / 128.0f);
    float v2 = 0.0f;
    for (int c = 0; c < 128; ++c) { float d0 = rp[c] - mu; v2 += d0 * d0; }
    float rs = rsqrtf(v2 * (1.0f / 128.0f) + 1e-5f);
    for (int c = 0; c < 128; ++c) wp[c] = f2bf((rp[c] - mu) * rs * sng[c] + snb[c]);
    if (m0 + lane < M) atomicAdd(&cnt[sind[m0 + lane]], 1.0f);
  }
  __syncthreads();

  int m = lane & 15, hi = lane >> 4;
  v16bf a[4];
#pragma unroll
  for (int ks = 0; ks < 4; ++ks) a[ks] = load_a_lds(&xb[wave][0], 128, lane, ks);

#pragma unroll
  for (int nt = 0; nt < 8; ++nt) {                // GEMM1 + SiLU -> LDS bf16
    v8f c = {};
#pragma unroll
    for (int ks = 0; ks < 4; ++ks) {
      v16bf b = load_b_glb(p1T, 128, lane, nt, ks);
      c = WMMA_BF16(a[ks], b, c);
    }
    int col = nt * 16 + m;
    float bb = b1[col];
#pragma unroll
    for (int r = 0; r < 8; ++r) {
      float hv = c[r] + bb;
      hv = hv / (1.0f + __expf(-hv));
      hb[wave][(r + 8 * hi) * 128 + col] = f2bf(hv);
    }
  }
  __syncthreads();

#pragma unroll
  for (int ks = 0; ks < 4; ++ks) a[ks] = load_a_lds(&hb[wave][0], 128, lane, ks);
#pragma unroll
  for (int nt = 0; nt < 8; ++nt) {                // GEMM2 + scatter-add
    v8f c = {};
#pragma unroll
    for (int ks = 0; ks < 4; ++ks) {
      v16bf b = load_b_glb(p2T, 128, lane, nt, ks);
      c = WMMA_BF16(a[ks], b, c);
    }
    int col = nt * 16 + m;
    float bb = b2[col];
    if (fullw) {
#pragma unroll
      for (int r = 0; r < 8; ++r) {
        int node = sind[m0 + r + 8 * hi];
        atomicAdd(&subacc[node * 128 + col], c[r] + bb);
      }
    } else {
#pragma unroll
      for (int r = 0; r < 8; ++r) {
        int ri = m0 + r + 8 * hi;
        if (ri < M) {
          int node = sind[ri];
          atomicAdd(&subacc[node * 128 + col], c[r] + bb);
        }
      }
    }
  }
}

// ---------------------------------------------------------------------------
// Finalize: out = silu(LN(x + attn + skip + subacc/cnt)). One wave per node.
__global__ __launch_bounds__(256) void finalize_kernel(
    const float* __restrict__ x, const float* __restrict__ skipb,
    const float* __restrict__ numb, const float* __restrict__ denb,
    const float* __restrict__ subacc, const float* __restrict__ cnt,
    const float* __restrict__ ong, const float* __restrict__ onb,
    float* __restrict__ out, int n) {
  int tid = threadIdx.x, wave = tid >> 5, lane = tid & 31;
  int row = blockIdx.x * 8 + wave;
  bool ok = row < n;
  int r2 = ok ? row : 0;
  int c0 = lane * 4;
  float4 xr = *reinterpret_cast<const float4*>(x + r2 * 128 + c0);
  float4 sk = *reinterpret_cast<const float4*>(skipb + r2 * 128 + c0);
  float4 nm = *reinterpret_cast<const float4*>(numb + r2 * 128 + c0);
  float4 sb = *reinterpret_cast<const float4*>(subacc + r2 * 128 + c0);
  float den = denb[r2 * 8 + (c0 >> 4)];
  den = (den > 0.0f) ? den : 1.0f;
  float inv = 1.0f / fmaxf(cnt[r2], 1.0f);
  float t0 = xr.x + sk.x + nm.x / den + sb.x * inv;
  float t1 = xr.y + sk.y + nm.y / den + sb.y * inv;
  float t2 = xr.z + sk.z + nm.z / den + sb.z * inv;
  float t3 = xr.w + sk.w + nm.w / den + sb.w * inv;
  float s = t0 + t1 + t2 + t3;
  float ss = t0 * t0 + t1 * t1 + t2 * t2 + t3 * t3;
#pragma unroll
  for (int msk = 1; msk < 32; msk <<= 1) {
    s += __shfl_xor(s, msk, 32);
    ss += __shfl_xor(ss, msk, 32);
  }
  float mu = s * (1.0f / 128.0f);
  float var = ss * (1.0f / 128.0f) - mu * mu;
  float rs = rsqrtf(var + 1e-5f);
  float4 g = *reinterpret_cast<const float4*>(ong + c0);
  float4 b = *reinterpret_cast<const float4*>(onb + c0);
  float o0 = (t0 - mu) * rs * g.x + b.x;
  float o1 = (t1 - mu) * rs * g.y + b.y;
  float o2 = (t2 - mu) * rs * g.z + b.z;
  float o3 = (t3 - mu) * rs * g.w + b.w;
  o0 = o0 / (1.0f + __expf(-o0));
  o1 = o1 / (1.0f + __expf(-o1));
  o2 = o2 / (1.0f + __expf(-o2));
  o3 = o3 / (1.0f + __expf(-o3));
  if (ok) {
    float4 o = {o0, o1, o2, o3};
    *reinterpret_cast<float4*>(out + row * 128 + c0) = o;
  }
}

// ---------------------------------------------------------------------------
extern "C" void kernel_launch(void* const* d_in, const int* in_sizes, int n_in,
                              void* d_out, int out_size, void* d_ws, size_t ws_size,
                              hipStream_t stream) {
  const float* x      = (const float*)d_in[0];
  const int*   ei     = (const int*)d_in[1];
  const float* eattr  = (const float*)d_in[2];
  const int*   snodes = (const int*)d_in[3];
  const int*   sind   = (const int*)d_in[4];
  const float* Wq = (const float*)d_in[5];  const float* bq = (const float*)d_in[6];
  const float* Wk = (const float*)d_in[7];  const float* bk = (const float*)d_in[8];
  const float* Wv = (const float*)d_in[9];  const float* bv = (const float*)d_in[10];
  const float* We = (const float*)d_in[11]; const float* be = (const float*)d_in[12];
  const float* Ws = (const float*)d_in[13]; const float* bs = (const float*)d_in[14];
  const float* sng = (const float*)d_in[15]; const float* snb = (const float*)d_in[16];
  const float* p1w = (const float*)d_in[17]; const float* p1b = (const float*)d_in[18];
  const float* p2w = (const float*)d_in[19]; const float* p2b = (const float*)d_in[20];
  const float* ong = (const float*)d_in[21]; const float* onb = (const float*)d_in[22];

  const int N = in_sizes[0] / 128;
  const int E = in_sizes[1] / 2;
  const int M = in_sizes[3];

  // ---- workspace layout (bytes) ----
  const size_t SZ_W  = 212992;                  // bf16 weights
  const size_t SZ_ND = (size_t)N * 128 * 4;
  const size_t SZ_NH = (size_t)N * 8 * 4;
  const size_t SZ_AL = (size_t)E * 8 * 4;
  const size_t OFF_Q   = SZ_W;
  const size_t OFF_K   = OFF_Q + SZ_ND;
  const size_t OFF_V   = OFF_K + SZ_ND;
  const size_t OFF_S   = OFF_V + SZ_ND;
  const size_t OFF_AL  = OFF_S + SZ_ND;
  const size_t OFF_MK  = OFF_AL + SZ_AL;        // start of zero-init region
  const size_t OFF_DEN = OFF_MK + SZ_NH;
  const size_t OFF_NUM = OFF_DEN + SZ_NH;
  const size_t OFF_SUB = OFF_NUM + SZ_ND;
  const size_t OFF_CNT = OFF_SUB + SZ_ND;
  const size_t END     = OFF_CNT + (size_t)N * 4;
  if (ws_size < END) return;                    // insufficient scratch: no-op

  char* wsb = (char*)d_ws;
  unsigned short* wts = (unsigned short*)wsb;
  float* qb  = (float*)(wsb + OFF_Q);
  float* kb  = (float*)(wsb + OFF_K);
  float* vb  = (float*)(wsb + OFF_V);
  float* sbk = (float*)(wsb + OFF_S);
  float* alp = (float*)(wsb + OFF_AL);
  unsigned* mk = (unsigned*)(wsb + OFF_MK);
  float* den = (float*)(wsb + OFF_DEN);
  float* num = (float*)(wsb + OFF_NUM);
  float* sub = (float*)(wsb + OFF_SUB);
  float* cnt = (float*)(wsb + OFF_CNT);
  const unsigned short* weT = wts + 65536;
  const unsigned short* p1T = wts + 73728;
  const unsigned short* p2T = wts + 90112;

  long zeroCount = (long)((END - OFF_MK) / 4);
  zero_kernel<<<2048, 256, 0, stream>>>((float*)(wsb + OFF_MK), zeroCount);
  prep_weights_kernel<<<128, 256, 0, stream>>>(Wq, Wk, Wv, Ws, We, p1w, p2w, wts);

  node_proj_kernel<<<(N + 63) / 64, 128, 0, stream>>>(
      x, wts, bq, bk, bv, bs, qb, kb, vb, sbk, N);

  edge_attn_a_kernel<<<(E + 63) / 64, 128, 0, stream>>>(
      eattr, ei, ei + E, weT, be, qb, kb, alp, mk, E);

  edge_attn_b_kernel<<<(E + 63) / 64, 128, 0, stream>>>(
      eattr, ei, ei + E, weT, be, vb, alp, mk, num, den, E);

  subgraph_kernel<<<(M + 63) / 64, 128, 0, stream>>>(
      x, snodes, sind, sng, snb, p1T, p1b, p2T, p2b, sub, cnt, M);

  finalize_kernel<<<(N + 7) / 8, 256, 0, stream>>>(
      x, sbk, num, den, sub, cnt, ong, onb, (float*)d_out, N);
}
